// SocialRelationalAttention_27496380629538
// MI455X (gfx1250) — matrix operand order repaired and checked by reading, not compiled, not run on hardware
//
#include <hip/hip_runtime.h>
#include <hip/hip_bf16.h>

// Shapes from the reference
#define TT 30
#define BB 32
#define NN 128
#define FF 128
#define HH 8
// DH = 16

typedef __attribute__((ext_vector_type(16))) __bf16 v16bf;
typedef __attribute__((ext_vector_type(8)))  __bf16 v8bf;
typedef __attribute__((ext_vector_type(4)))  __bf16 v4bf;
typedef __attribute__((ext_vector_type(8)))  float  v8f;

static constexpr int LS = 136;  // bf16 LDS row stride (272B = 17*16, keeps 16B alignment, breaks bank conflicts)
static constexpr int RS = 132;  // f32 residual row stride

// Native conversions: let the compiler emit v_cvt_*_bf16 (hw RNE) instead of bit-twiddling
__device__ __forceinline__ __bf16 f2bf(float f) { return static_cast<__bf16>(f); }
__device__ __forceinline__ float bf2f(__bf16 b) { return static_cast<float>(b); }

__device__ __forceinline__ float logsig(float v) {
  // log(sigmoid(v)) = -softplus(-v), numerically stable split
  float l = -__logf(1.0f + __expf(-fabsf(v)));
  return v < 0.0f ? v + l : l;
}
__device__ __forceinline__ v8f wmma_bf16(v16bf a, v16bf b, v8f c) {
  return __builtin_amdgcn_wmma_f32_16x16x32_bf16(false, a, false, b, (short)0, c, false, false);
}

// A fragment (16x32 bf16): lane l -> row rowBase+(l&15); K = k0 + (l>>4)*16 + j
__device__ __forceinline__ v16bf loadA(const __bf16* M, int rowBase, int k0, int lane) {
  int m = lane & 15, h = lane >> 4;
  const __bf16* p0 = M + (rowBase + m) * LS + k0 + h * 16;
  v8bf lo = *(const v8bf*)p0;
  v8bf hi = *(const v8bf*)(p0 + 8);
  v16bf a;
#pragma unroll
  for (int j = 0; j < 8; ++j) { a[j] = lo[j]; a[8 + j] = hi[j]; }
  return a;
}

// B fragment (32x16 bf16) from row-major [outIdx][K] LDS: lane l -> out col nBase+(l&15); K = k0 + (l>>4)*16 + j
__device__ __forceinline__ v16bf loadB(const __bf16* M, int nBase, int k0, int lane) {
  int n = lane & 15, h = lane >> 4;
  const __bf16* p0 = M + (nBase + n) * LS + k0 + h * 16;
  v8bf lo = *(const v8bf*)p0;
  v8bf hi = *(const v8bf*)(p0 + 8);
  v16bf b;
#pragma unroll
  for (int j = 0; j < 8; ++j) { b[j] = lo[j]; b[8 + j] = hi[j]; }
  return b;
}

// B fragment from global f32 weight matrix [128][128] (row-major, K contiguous), converted to bf16 in-flight
__device__ __forceinline__ v16bf loadBW(const float* __restrict__ W, int nBase, int k0, int lane) {
  int n = lane & 15, h = lane >> 4;
  const float* p = W + (nBase + n) * FF + k0 + h * 16;
  v16bf b;
#pragma unroll
  for (int j = 0; j < 16; j += 4) {
    float4 f = *(const float4*)(p + j);
    b[j] = f2bf(f.x); b[j + 1] = f2bf(f.y); b[j + 2] = f2bf(f.z); b[j + 3] = f2bf(f.w);
  }
  return b;
}

__global__ __launch_bounds__(256, 1)
void sra_fused_kernel(const float* __restrict__ input, const float* __restrict__ ipos,
                      const unsigned char* __restrict__ mask,
                      const float* __restrict__ Wq, const float* __restrict__ bq,
                      const float* __restrict__ Wk, const float* __restrict__ bk,
                      const float* __restrict__ Wv, const float* __restrict__ bv,
                      const float* __restrict__ Wr, const float* __restrict__ br,
                      const float* __restrict__ Wc, const float* __restrict__ bc,
                      float* __restrict__ out) {
  extern __shared__ char smem[];
  float*  res  = (float*)smem;                              // [128][132] f32  (input - pos)
  __bf16* xbf  = (__bf16*)(smem + NN * RS * 4);             // [128][136] bf16 (x = input + pos); reused as p-strips
  __bf16* qbf  = xbf  + NN * LS;                            // [128][136] q * 0.25
  __bf16* kbf  = qbf  + NN * LS;                            // [128][136] k
  __bf16* lsvT = kbf  + NN * LS;                            // [128][136] log_sigmoid(v) TRANSPOSED: [dim][token]
  __bf16* rbf  = lsvT + NN * LS;                            // [128][136] r
  __bf16* obf  = rbf  + NN * LS;                            // [128][136] r * exp(att)
  float*  madd = (float*)(obf + NN * LS);                   // [128] column mask add (0 or -1e9)

  const int tid  = threadIdx.x;
  const int lane = tid & 31;
  const int wid  = tid >> 5;          // 8 waves
  const int bx   = blockIdx.x;        // t*B + b
  const int bb   = bx & (BB - 1);

  const float* xin = input + (size_t)bx * (NN * FF);
  const float* pin = ipos  + (size_t)bx * (NN * FF);
  float*       og  = out   + (size_t)bx * (NN * FF);

  // ---------------- Stage A: x (bf16), residual (f32), column mask ----------------
  for (int i = tid; i < NN * FF / 4; i += 256) {
    float4 a = ((const float4*)xin)[i];
    float4 c = ((const float4*)pin)[i];
    int row = i >> 5, cc = (i & 31) * 4;
    float4 r4; r4.x = a.x - c.x; r4.y = a.y - c.y; r4.z = a.z - c.z; r4.w = a.w - c.w;
    *(float4*)(res + row * RS + cc) = r4;
    v4bf xb;
    xb[0] = f2bf(a.x + c.x); xb[1] = f2bf(a.y + c.y);
    xb[2] = f2bf(a.z + c.z); xb[3] = f2bf(a.w + c.w);
    *(v4bf*)(xbf + row * LS + cc) = xb;
  }
  if (tid < NN) {
    unsigned char any = 0;
    const unsigned char* mp = mask + (size_t)bb * NN + tid;
    for (int t = 0; t < TT; ++t) any |= mp[(size_t)t * BB * NN];
    madd[tid] = any ? 0.0f : -1e9f;
  }
  __syncthreads();

  // ---------------- Stage B: projections q,k,v,r (wave w owns out-cols [16w,16w+16)) ----------------
  {
    const float* Wp[4] = {Wq, Wk, Wv, Wr};
    const float* bp[4] = {bq, bk, bv, br};
    const int ni = wid;
    const int c0 = lane & 15;
#pragma unroll
    for (int p = 0; p < 4; ++p) {
      v16bf bfr[4];
#pragma unroll
      for (int k = 0; k < 4; ++k) bfr[k] = loadBW(Wp[p], ni * 16, k * 32, lane);
      float bias = bp[p][ni * 16 + c0];
      for (int mi = 0; mi < 8; ++mi) {
        v8f acc = {};
#pragma unroll
        for (int k = 0; k < 4; ++k) acc = wmma_bf16(loadA(xbf, mi * 16, k * 32, lane), bfr[k], acc);
        int rbase = mi * 16 + (lane >> 4) * 8;
        int col = ni * 16 + c0;
#pragma unroll
        for (int i = 0; i < 8; ++i) {
          float v = acc[i] + bias;
          int row = rbase + i;
          if      (p == 0) qbf[row * LS + col] = f2bf(v * 0.25f);      // fold 1/sqrt(DH)
          else if (p == 1) kbf[row * LS + col] = f2bf(v);
          else if (p == 2) lsvT[col * LS + row] = f2bf(logsig(v));     // transposed store
          else             rbf[row * LS + col] = f2bf(v);
        }
      }
    }
  }
  __syncthreads();

  // ---------------- Stage C: attention, wave w == head w ----------------
  {
    const int h  = wid;
    const int c0 = lane & 15;
    const int rl = (lane >> 4) * 8;
    __bf16* ps = xbf + h * 16 * LS;            // wave-private 16x136 p-strip (xbf is dead)
    float mcol[8];
#pragma unroll
    for (int ni = 0; ni < 8; ++ni) mcol[ni] = madd[ni * 16 + c0];

    for (int mi = 0; mi < 8; ++mi) {
      // A fragment of q for this row strip: K 0..15 = head dims, K 16..31 zero-padded
      v16bf aq = {};
      if (lane < 16) {
        const __bf16* p0 = qbf + (mi * 16 + c0) * LS + h * 16;
        v8bf lo = *(const v8bf*)p0;
        v8bf hi = *(const v8bf*)(p0 + 8);
#pragma unroll
        for (int j = 0; j < 8; ++j) { aq[j] = lo[j]; aq[8 + j] = hi[j]; }
      }
      v8f s[8];
#pragma unroll
      for (int ni = 0; ni < 8; ++ni) {
        v16bf bk_ = {};
        if (lane < 16) {                       // lanes 16..31 carry K=16..31 which A zeroes out
          const __bf16* p0 = kbf + (ni * 16 + lane) * LS + h * 16;
          v8bf lo = *(const v8bf*)p0;
          v8bf hi = *(const v8bf*)(p0 + 8);
#pragma unroll
          for (int j = 0; j < 8; ++j) { bk_[j] = lo[j]; bk_[8 + j] = hi[j]; }
        }
        v8f z = {};
        s[ni] = wmma_bf16(aq, bk_, z);
      }
      // mask + row softmax (row = mi*16 + rl + i, its 128 cols live in 8 accs x 16 lanes of this half)
      float mx[8], sm[8];
#pragma unroll
      for (int i = 0; i < 8; ++i) {
        float m = -3.4e38f;
#pragma unroll
        for (int ni = 0; ni < 8; ++ni) {
          float v = (mcol[ni] == 0.0f) ? s[ni][i] : -1e9f;
          s[ni][i] = v;
          m = fmaxf(m, v);
        }
        mx[i] = m;
      }
      for (int d = 1; d < 16; d <<= 1)
#pragma unroll
        for (int i = 0; i < 8; ++i) mx[i] = fmaxf(mx[i], __shfl_xor(mx[i], d, 16));
#pragma unroll
      for (int i = 0; i < 8; ++i) {
        float t = 0.0f;
#pragma unroll
        for (int ni = 0; ni < 8; ++ni) { float e = __expf(s[ni][i] - mx[i]); s[ni][i] = e; t += e; }
        sm[i] = t;
      }
      for (int d = 1; d < 16; d <<= 1)
#pragma unroll
        for (int i = 0; i < 8; ++i) sm[i] += __shfl_xor(sm[i], d, 16);
#pragma unroll
      for (int i = 0; i < 8; ++i) sm[i] = 1.0f / sm[i];
      // write normalized probabilities to wave-private strip (bf16, A-matmul layout)
#pragma unroll
      for (int ni = 0; ni < 8; ++ni)
#pragma unroll
        for (int i = 0; i < 8; ++i)
          ps[(rl + i) * LS + ni * 16 + c0] = f2bf(s[ni][i] * sm[i]);

      // att = p[16,128] @ lsv_h[128,16] (same-wave LDS RAW -> compiler s_wait_dscnt)
      v8f att = {};
#pragma unroll
      for (int k = 0; k < 4; ++k)
        att = wmma_bf16(loadA(ps, 0, k * 32, lane), loadB(lsvT, h * 16, k * 32, lane), att);

      // out_h = r * exp(att)
#pragma unroll
      for (int i = 0; i < 8; ++i) {
        int row = mi * 16 + rl + i;
        int col = h * 16 + c0;
        float o = bf2f(rbf[row * LS + col]) * __expf(att[i]);
        obf[row * LS + col] = f2bf(o);
      }
    }
  }
  __syncthreads();

  // ---------------- Stage D: final projection + bias + residual ----------------
  {
    const int ni = wid;
    const int c0 = lane & 15;
    v16bf bw[4];
#pragma unroll
    for (int k = 0; k < 4; ++k) bw[k] = loadBW(Wc, ni * 16, k * 32, lane);
    float bias = bc[ni * 16 + c0];
    for (int mi = 0; mi < 8; ++mi) {
      v8f acc = {};
#pragma unroll
      for (int k = 0; k < 4; ++k) acc = wmma_bf16(loadA(obf, mi * 16, k * 32, lane), bw[k], acc);
      int rbase = mi * 16 + (lane >> 4) * 8;
      int col = ni * 16 + c0;
#pragma unroll
      for (int i = 0; i < 8; ++i) {
        int row = rbase + i;
        og[row * FF + col] = acc[i] + bias + res[row * RS + col];
      }
    }
  }
}

static constexpr size_t SMEM_BYTES =
    (size_t)NN * RS * 4 + (size_t)6 * NN * LS * 2 + NN * 4;  // 67584 + 208896 + 512 = 276992

extern "C" void kernel_launch(void* const* d_in, const int* in_sizes, int n_in,
                              void* d_out, int out_size, void* d_ws, size_t ws_size,
                              hipStream_t stream) {
  (void)in_sizes; (void)n_in; (void)out_size; (void)d_ws; (void)ws_size;
  const float* input = (const float*)d_in[0];
  const float* ipos  = (const float*)d_in[1];
  const unsigned char* mask = (const unsigned char*)d_in[2];
  const float* Wq = (const float*)d_in[3];  const float* bq = (const float*)d_in[4];
  const float* Wk = (const float*)d_in[5];  const float* bk = (const float*)d_in[6];
  const float* Wv = (const float*)d_in[7];  const float* bv = (const float*)d_in[8];
  const float* Wr = (const float*)d_in[9];  const float* br = (const float*)d_in[10];
  const float* Wc = (const float*)d_in[11]; const float* bc = (const float*)d_in[12];
  float* out = (float*)d_out;

  hipFuncSetAttribute((const void*)sra_fused_kernel,
                      hipFuncAttributeMaxDynamicSharedMemorySize, (int)SMEM_BYTES);
  sra_fused_kernel<<<dim3(TT * BB), dim3(256), SMEM_BYTES, stream>>>(
      input, ipos, mask, Wq, bq, Wk, bk, Wv, bv, Wr, br, Wc, bc, out);
}
